// TensorConvolutionTrainLayer_45913200394853
// MI455X (gfx1250) — matrix operand order repaired and checked by reading, not compiled, not run on hardware
//
#include <hip/hip_runtime.h>

// ---- problem dims ----
#define S_DIM 512
#define P_DIM 196
#define Q_DIM 48
#define CB_DIM 8
#define R_DIM 32
#define C_DIM 10
#define N_STEPS 6

#define AB 64           // (a,b) fused CB x CB
#define LR 1024         // (l,r) fused R x R
#define AL 256          // (a,l) fused CB x R
#define BR 256          // (b,r) fused CB x R
#define PPAD 224        // P padded to 7*32 for K-chunks
#define QPAD 64         // Q padded to 2*32
#define STATE_SZ (AL * C_DIM)   // 2560 f32 per sample

// ---- workspace layout (bytes) ----
#define WS_CKT  0ull                                   // f16 [6][AB][QPAD]      = 49152 B
#define WS_AKT  49152ull                               // f16 [6][LR][PPAD]      = 2752512 B
#define WS_STA  (49152ull + 2752512ull)                // f32 [S][AL][C]         = 5242880 B
#define WS_STB  (WS_STA + 5242880ull)

// ---- dynamic LDS layout for the step kernel ----
// region0: phases 0-A: Xs[208][64] f16 (26624) + CkTs[64][64] f16 (@26624, 8192)
//          phases B-C: GhatT[256][256] f16 (131072)  -- aliases Xs/CkTs (dead by then)
// MsT[64][224] f16 @131072 (28672) ; AT[16][256] f16 aliases MsT after phase B.
#define SM_CKT   26624
#define SM_MST   131072
#define SM_AT    131072
#define SM_TOTAL 159744

typedef __attribute__((ext_vector_type(16))) _Float16 v16h;
typedef __attribute__((ext_vector_type(8)))  _Float16 v8h;
typedef __attribute__((ext_vector_type(8)))  float    v8f;

// ============ prep: permute + pad weights into f16 GEMM layouts ============
__global__ void __launch_bounds__(256) tctl_prep(const float* __restrict__ conv_mid,
                                                 const float* __restrict__ train_mid,
                                                 _Float16* __restrict__ CkT,
                                                 _Float16* __restrict__ AkT) {
  const size_t total1 = (size_t)N_STEPS * AB * QPAD;   // 24576
  const size_t total2 = (size_t)N_STEPS * LR * PPAD;   // 1376256
  for (size_t idx = (size_t)blockIdx.x * blockDim.x + threadIdx.x;
       idx < total1 + total2; idx += (size_t)gridDim.x * blockDim.x) {
    if (idx < total1) {
      size_t k = idx / (AB * QPAD), rem = idx % (AB * QPAD);
      int ab = (int)(rem / QPAD), q = (int)(rem % QPAD);
      int a = ab >> 3, b = ab & 7;
      float v = (q < Q_DIM)
          ? conv_mid[(((size_t)k * CB_DIM + a) * Q_DIM + q) * CB_DIM + b] : 0.f;
      CkT[idx] = (_Float16)v;
    } else {
      size_t j = idx - total1;
      size_t k = j / ((size_t)LR * PPAD), rem = j % ((size_t)LR * PPAD);
      int lr = (int)(rem / PPAD), p = (int)(rem % PPAD);
      int l = lr >> 5, r = lr & 31;
      float v = (p < P_DIM)
          ? train_mid[(((size_t)k * R_DIM + l) * P_DIM + p) * R_DIM + r] : 0.f;
      AkT[j] = (_Float16)v;
    }
  }
}

// ============ first carriage: state0[s][b*32+r][c] ============
__global__ void __launch_bounds__(256) tctl_init(const float* __restrict__ x,
                                                 const float* __restrict__ cf,
                                                 const float* __restrict__ tf,
                                                 float* __restrict__ st_out) {
  __shared__ float Y[P_DIM * CB_DIM];
  const int tid = threadIdx.x, s = blockIdx.x;
  for (int i = tid; i < P_DIM * CB_DIM; i += 256) {
    int p = i >> 3, b = i & 7;
    const float* xr = &x[((size_t)s * P_DIM + p) * Q_DIM];
    float acc = 0.f;
    for (int q = 0; q < Q_DIM; ++q) acc += xr[q] * cf[q * CB_DIM + b];
    Y[i] = acc;
  }
  __syncthreads();
  for (int i = tid; i < STATE_SZ; i += 256) {
    int c = i % C_DIM, br = i / C_DIM;
    int b = br >> 5, r = br & 31;
    float acc = 0.f;
    for (int p = 0; p < P_DIM; ++p)
      acc += Y[p * CB_DIM + b] * tf[((size_t)c * P_DIM + p) * R_DIM + r];
    st_out[(size_t)s * STATE_SZ + i] = acc;
  }
}

// ============ mid carriage: the WMMA kernel (one workgroup per sample) ============
__global__ void __launch_bounds__(256) tctl_step(const float* __restrict__ x,
                                                 const _Float16* __restrict__ CkT,  // [AB][QPAD]
                                                 const _Float16* __restrict__ AkT,  // [LR][PPAD]
                                                 const float* __restrict__ st_in,
                                                 float* __restrict__ st_out) {
  extern __shared__ __align__(32) char smem[];
  _Float16* Xs    = (_Float16*)(smem);            // [208][64]
  _Float16* CkTs  = (_Float16*)(smem + SM_CKT);   // [64][64]
  _Float16* GhatT = (_Float16*)(smem);            // [BR][AL], aliases Xs/CkTs
  _Float16* MsT   = (_Float16*)(smem + SM_MST);   // [64][224]
  _Float16* AT    = (_Float16*)(smem + SM_AT);    // [16][256], aliases MsT

  const int tid  = threadIdx.x;
  const int lane = tid & 31;
  const int wid  = tid >> 5;
  const int s    = blockIdx.x;
  const int half = lane >> 4;     // 0/1 : K-half select per CDNA5 wave32 layouts
  const int l15  = lane & 15;

  // --- phase 0: stage x_s (f32->f16, zero-padded) + CkT tile; zero MsT pad cols ---
  for (int i = tid; i < 208 * 64; i += 256) {
    int p = i >> 6, q = i & 63;
    float v = (p < P_DIM && q < Q_DIM) ? x[((size_t)s * P_DIM + p) * Q_DIM + q] : 0.f;
    Xs[i] = (_Float16)v;
  }
  for (int i = tid; i < AB * QPAD; i += 256) CkTs[i] = CkT[i];
  for (int i = tid; i < AB * PPAD; i += 256) MsT[i] = (_Float16)0.f;
  __syncthreads();

  // --- phase A: M[p,ab] = Xs(208x64) x Ck  -> store transposed MsT[ab][p] ---
  for (int t = wid; t < 13 * 4; t += 8) {
    int it = t >> 2, jt = t & 3;
    v8f acc = {};
#pragma unroll
    for (int kc = 0; kc < 2; ++kc) {
      int m = it * 16 + l15;
      int klo = kc * 32 + (half << 3);
      v16h a;
      ((v8h*)&a)[0] = *(const v8h*)&Xs[m * 64 + klo];
      ((v8h*)&a)[1] = *(const v8h*)&Xs[m * 64 + klo + 16];
      int n = jt * 16 + l15;
      int kb = kc * 32 + (half << 3);
      v16h b;
      ((v8h*)&b)[0] = *(const v8h*)&CkTs[n * 64 + kb];
      ((v8h*)&b)[1] = *(const v8h*)&CkTs[n * 64 + kb + 16];
      acc = __builtin_amdgcn_wmma_f32_16x16x32_f16(false, a, false, b, (short)0, acc,
                                                   false, false);
    }
    for (int j = 0; j < 8; ++j) {
      int p  = it * 16 + j + (half << 3);
      int ab = jt * 16 + l15;
      MsT[ab * PPAD + p] = (_Float16)acc[j];
    }
  }
  __syncthreads();

  // --- phase B: G[ab,lr] = MsT(64x224) x AkT^T ---
  // Wave w owns row-tile it = w>>1 and N-tiles jt in [(w&1)*32, (w&1)*32+32).
  // A fragments (7 K-chunks) are invariant over the jt sweep -> load once into regs.
  {
    const int it  = wid >> 1;
    const int jt0 = (wid & 1) * 32;
    const int m   = it * 16 + l15;
    v16h afrag[7];
#pragma unroll
    for (int kc = 0; kc < 7; ++kc) {
      int klo = kc * 32 + (half << 3);
      ((v8h*)&afrag[kc])[0] = *(const v8h*)&MsT[m * PPAD + klo];
      ((v8h*)&afrag[kc])[1] = *(const v8h*)&MsT[m * PPAD + klo + 16];
    }
    for (int jt = jt0; jt < jt0 + 32; ++jt) {
      int n = jt * 16 + l15;
      const _Float16* brow = &AkT[(size_t)n * PPAD];
      if (jt + 1 < jt0 + 32)
        __builtin_prefetch((const void*)(brow + 16 * PPAD), 0, 1);  // next tile's B row
      v8f acc = {};
#pragma unroll
      for (int kc = 0; kc < 7; ++kc) {
        int kb = kc * 32 + (half << 3);
        v16h b;
        ((v8h*)&b)[0] = *(const v8h*)&brow[kb];
        ((v8h*)&b)[1] = *(const v8h*)&brow[kb + 16];
        acc = __builtin_amdgcn_wmma_f32_16x16x32_f16(false, afrag[kc], false, b,
                                                     (short)0, acc, false, false);
      }
      // scatter, pre-transposed: GhatT[(b*32+r)][(a*32+l)]
#pragma unroll
      for (int j = 0; j < 8; ++j) {
        int ab = it * 16 + j + (half << 3);
        int lr = jt * 16 + l15;
        int aa = ab >> 3, bb = ab & 7, ll = lr >> 5, rr = lr & 31;
        GhatT[(bb * 32 + rr) * AL + (aa * 32 + ll)] = (_Float16)acc[j];
      }
    }
  }
  __syncthreads();   // GhatT complete; MsT now dead -> AT may overwrite it

  // --- stage state^T: AT[c][al], c padded to 16 ---
  for (int i = tid; i < 16 * AL; i += 256) {
    int c = i >> 8, al = i & 255;
    float v = (c < C_DIM) ? st_in[(size_t)s * STATE_SZ + al * C_DIM + c] : 0.f;
    AT[i] = (_Float16)v;
  }
  __syncthreads();

  // --- phase C: new_state[c][(b,r)] = AT(16x256) x Ghat(256x256) ---
  // B operand: column (b,r)=n fixed, K=(a,l) contiguous -> contiguous row of GhatT.
  for (int t = wid; t < 16; t += 8) {
    int n = t * 16 + l15;
    v8f acc = {};
#pragma unroll
    for (int kc = 0; kc < 8; ++kc) {
      int klo = kc * 32 + (half << 3);
      v16h a;
      ((v8h*)&a)[0] = *(const v8h*)&AT[l15 * AL + klo];
      ((v8h*)&a)[1] = *(const v8h*)&AT[l15 * AL + klo + 16];
      int kb = kc * 32 + (half << 3);
      v16h b;
      ((v8h*)&b)[0] = *(const v8h*)&GhatT[n * AL + kb];
      ((v8h*)&b)[1] = *(const v8h*)&GhatT[n * AL + kb + 16];
      acc = __builtin_amdgcn_wmma_f32_16x16x32_f16(false, a, false, b, (short)0, acc,
                                                   false, false);
    }
    for (int j = 0; j < 8; ++j) {
      int c = j + (half << 3);
      if (c < C_DIM)
        st_out[(size_t)s * STATE_SZ + n * C_DIM + c] = acc[j];
    }
  }
}

// ============ last carriage: out[s][c] ============
__global__ void __launch_bounds__(256) tctl_final(const float* __restrict__ x,
                                                  const float* __restrict__ cl,
                                                  const float* __restrict__ tl,
                                                  const float* __restrict__ st,
                                                  float* __restrict__ out) {
  __shared__ float Y2[P_DIM * CB_DIM];
  __shared__ float GN[AL];
  const int tid = threadIdx.x, s = blockIdx.x;
  for (int i = tid; i < P_DIM * CB_DIM; i += 256) {
    int p = i >> 3, a = i & 7;
    const float* xr = &x[((size_t)s * P_DIM + p) * Q_DIM];
    float acc = 0.f;
    for (int q = 0; q < Q_DIM; ++q) acc += xr[q] * cl[a * Q_DIM + q];
    Y2[i] = acc;
  }
  __syncthreads();
  for (int i = tid; i < AL; i += 256) {
    int a = i >> 5, l = i & 31;
    float acc = 0.f;
    for (int p = 0; p < P_DIM; ++p) acc += Y2[p * CB_DIM + a] * tl[l * P_DIM + p];
    GN[i] = acc;
  }
  __syncthreads();
  if (tid < C_DIM) {
    float acc = 0.f;
    for (int al = 0; al < AL; ++al)
      acc += st[(size_t)s * STATE_SZ + al * C_DIM + tid] * GN[al];
    out[(size_t)s * C_DIM + tid] = acc;
  }
}

extern "C" void kernel_launch(void* const* d_in, const int* in_sizes, int n_in,
                              void* d_out, int out_size, void* d_ws, size_t ws_size,
                              hipStream_t stream) {
  (void)in_sizes; (void)n_in; (void)out_size; (void)ws_size;
  const float* x  = (const float*)d_in[0];
  const float* cf = (const float*)d_in[1];
  const float* cm = (const float*)d_in[2];
  const float* cl = (const float*)d_in[3];
  const float* tf = (const float*)d_in[4];
  const float* tm = (const float*)d_in[5];
  const float* tl = (const float*)d_in[6];
  float* out = (float*)d_out;
  char* ws = (char*)d_ws;

  _Float16* CkT = (_Float16*)(ws + WS_CKT);
  _Float16* AkT = (_Float16*)(ws + WS_AKT);
  float* stA = (float*)(ws + WS_STA);
  float* stB = (float*)(ws + WS_STB);

  (void)hipFuncSetAttribute((const void*)tctl_step,
                            hipFuncAttributeMaxDynamicSharedMemorySize, SM_TOTAL);

  tctl_prep<<<512, 256, 0, stream>>>(cm, tm, CkT, AkT);
  tctl_init<<<S_DIM, 256, 0, stream>>>(x, cf, tf, stA);
  for (int k = 0; k < N_STEPS; ++k) {
    const float* sin = (k & 1) ? stB : stA;
    float*       sou = (k & 1) ? stA : stB;
    tctl_step<<<S_DIM, 256, SM_TOTAL, stream>>>(
        x, CkT + (size_t)k * AB * QPAD, AkT + (size_t)k * LR * PPAD, sin, sou);
  }
  // after 6 steps (even count) the live state is back in stA
  tctl_final<<<S_DIM, 256, 0, stream>>>(x, cl, tl, stA, out);
}
